// TaggingFNNCRFDecoder_25804163514906
// MI455X (gfx1250) — compile-verified, matrix operand order checked by reading, not run on hardware
//
#include <hip/hip_runtime.h>
#include <hip/hip_bf16.h>
#include <stdint.h>

typedef __bf16 bf16;
typedef __attribute__((ext_vector_type(16))) __bf16 v16bf;
typedef __attribute__((ext_vector_type(8)))  __bf16 v8bf;
typedef __attribute__((ext_vector_type(8)))  float  v8f;
typedef int v4i __attribute__((vector_size(16)));  // matches builtin's V4i param

#define LDT 40  // padded LDS row stride (bf16 elems): 80B rows, 16B aligned, spreads banks

// ---------------------------------------------------------------------------
// gfx1250 async global->LDS DMA (ASYNCcnt-tracked), guarded so either way the
// file compiles; fallback is plain global_load + ds_store staging.
// Builtin signature (from hipcc diagnostic): (v4i AS1*, v4i AS3*, Ii, Ii)
// ---------------------------------------------------------------------------
#if defined(__gfx1250__) && __has_builtin(__builtin_amdgcn_global_load_async_to_lds_b128)
#define HAVE_ASYNC 1
#else
#define HAVE_ASYNC 0
#endif

__device__ __forceinline__ void async_cp16(const bf16* g, bf16* l) {
#if HAVE_ASYNC
  __builtin_amdgcn_global_load_async_to_lds_b128(
      (__attribute__((address_space(1))) v4i*)g,
      (__attribute__((address_space(3))) v4i*)l, 0, 0);
#else
  *(v8bf*)l = *(const v8bf*)g;
#endif
}

__device__ __forceinline__ void wait_async0() {
#if HAVE_ASYNC
#if __has_builtin(__builtin_amdgcn_s_wait_asynccnt)
  __builtin_amdgcn_s_wait_asynccnt(0);
#else
  asm volatile("s_wait_asynccnt 0x0" ::: "memory");
#endif
#endif
}

// ---------------------------------------------------------------------------
// Fragment load from LDS following the CDNA5 16-bit A/B VGPR layout:
// lane 0-15  : row = lane,    K = {0..7, 16..23}
// lane 16-31 : row = lane-16, K = {8..15, 24..31}
// Two ds_load_b128 per fragment.
// ---------------------------------------------------------------------------
__device__ __forceinline__ v16bf load_frag(const bf16* tile, int lane) {
  const bf16* p = tile + (lane & 15) * LDT + ((lane >> 4) << 3);
  v8bf lo = *(const v8bf*)p;
  v8bf hi = *(const v8bf*)(p + 16);
  return __builtin_shufflevector(lo, hi, 0,1,2,3,4,5,6,7,8,9,10,11,12,13,14,15);
}

__device__ __forceinline__ v8f wmma_bf16(v16bf a, v16bf b, v8f c) {
  return __builtin_amdgcn_wmma_f32_16x16x32_bf16(false, a, false, b, (short)0, c,
                                                 false, false);
}

// ---------------------------------------------------------------------------
// Elementwise f32 -> bf16 convert
// ---------------------------------------------------------------------------
__global__ void cvt_f32_bf16(const float* __restrict__ in, bf16* __restrict__ out,
                             size_t n) {
  size_t i = (size_t)blockIdx.x * blockDim.x + threadIdx.x;
  size_t stride = (size_t)gridDim.x * blockDim.x;
  for (; i < n; i += stride) out[i] = (bf16)in[i];
}

__global__ void fill_zero_u32(uint32_t* __restrict__ p, size_t n) {
  size_t i = (size_t)blockIdx.x * blockDim.x + threadIdx.x;
  size_t stride = (size_t)gridDim.x * blockDim.x;
  for (; i < n; i += stride) p[i] = 0u;
}

// ---------------------------------------------------------------------------
// C[M,N] f32 = A[M,K]bf16 @ B[N,K]bf16^T + bias0[N] + bias1[N]
// Block = 128x128, 256 threads = 8 waves, wave tile 32x64 (2x4 WMMA tiles).
// LDS double-buffered staging (40KB static), async DMA when available.
// M must be a multiple of 128. BOUNDED=true adds N bounds checks (fc, N=74);
// the hot path (N=1024) is unbounded -> no exec-mask churn in the loop.
// ---------------------------------------------------------------------------
template <bool BOUNDED>
__global__ __launch_bounds__(256)
void gemm_bf16_nt(const bf16* __restrict__ A, int lda,
                  const bf16* __restrict__ Bw, int ldb,
                  const float* __restrict__ bias0,
                  const float* __restrict__ bias1,
                  float* __restrict__ C, int ldc,
                  int M, int N, int Kd) {
  __shared__ bf16 As[2][128 * LDT];
  __shared__ bf16 Bs[2][128 * LDT];

  const int m0 = blockIdx.x * 128;
  const int n0 = blockIdx.y * 128;
  const int tid = threadIdx.x;
  const int lane = tid & 31;
  const int wave = tid >> 5;
  const int wr = wave & 3;   // 32-row group
  const int wc = wave >> 2;  // 64-col group

  v8f acc[2][4];
#pragma unroll
  for (int i = 0; i < 2; ++i)
#pragma unroll
    for (int j = 0; j < 4; ++j)
#pragma unroll
      for (int v = 0; v < 8; ++v) acc[i][j][v] = 0.f;

  const int ksteps = Kd >> 5;

  auto stage = [&](int buf, int kt) {
#pragma unroll
    for (int i = 0; i < 2; ++i) {
      int cidx = tid + i * 256;  // 512 chunks of 8 bf16 per matrix
      int r = cidx >> 2;
      int kq = (cidx & 3) << 3;
      const bf16* ga = A + (size_t)(m0 + r) * lda + (kt << 5) + kq;
      async_cp16(ga, &As[buf][r * LDT + kq]);
      if (BOUNDED) {
        int brow = n0 + r;
        v8bf vb;
        if (brow < N) {
          vb = *(const v8bf*)(Bw + (size_t)brow * ldb + (kt << 5) + kq);
        } else {
#pragma unroll
          for (int e = 0; e < 8; ++e) vb[e] = (bf16)0.f;
        }
        *(v8bf*)&Bs[buf][r * LDT + kq] = vb;
      } else {
        const bf16* gb = Bw + (size_t)(n0 + r) * ldb + (kt << 5) + kq;
        async_cp16(gb, &Bs[buf][r * LDT + kq]);
      }
    }
  };

  stage(0, 0);
  wait_async0();
  __syncthreads();

  for (int kt = 0; kt < ksteps; ++kt) {
    int buf = kt & 1;
    if (kt + 1 < ksteps) stage(buf ^ 1, kt + 1);  // DMA overlaps WMMAs below
    v16bf af[2], bfg[4];
#pragma unroll
    for (int i = 0; i < 2; ++i)
      af[i] = load_frag(&As[buf][(wr * 32 + i * 16) * LDT], lane);
#pragma unroll
    for (int j = 0; j < 4; ++j)
      bfg[j] = load_frag(&Bs[buf][(wc * 64 + j * 16) * LDT], lane);
#pragma unroll
    for (int i = 0; i < 2; ++i)
#pragma unroll
      for (int j = 0; j < 4; ++j)
        acc[i][j] = wmma_bf16(af[i], bfg[j], acc[i][j]);
    wait_async0();
    __syncthreads();
  }

  // Epilogue: C/D layout -> lane<16 holds M=v, lane>=16 holds M=8+v; N = lane&15
#pragma unroll
  for (int i = 0; i < 2; ++i) {
    int rowBase = m0 + wr * 32 + i * 16 + ((lane >> 4) << 3);
#pragma unroll
    for (int j = 0; j < 4; ++j) {
      int col = n0 + wc * 64 + j * 16 + (lane & 15);
      if (!BOUNDED || col < N) {
        float bb = 0.f;
        if (bias0) bb += bias0[col];
        if (bias1) bb += bias1[col];
#pragma unroll
        for (int v = 0; v < 8; ++v)
          C[(size_t)(rowBase + v) * ldc + col] = acc[i][j][v] + bb;
      }
    }
  }
}

// ---------------------------------------------------------------------------
// One LSTM timestep, both directions fused (blockIdx.y = dir).
// Each block owns j-slice [j0, j0+32) across ALL 4 gates: it computes the
// 128(batch) x 128(4 gates x 32 j) tile of  z = h_prev @ w_hh^T, with w_hh
// rows remapped as  row(r) = (r>>5)*256 + j0 + (r&31).
// Wave tiling: wave w -> batch rows [w*16, w*16+16), full 128 cols
// (8 WMMA accumulators). For a fixed (b, jj) all 4 gate values live in the
// SAME lane / same accumulator element index -> register-only gate fusion.
// ---------------------------------------------------------------------------
__global__ __launch_bounds__(256)
void lstm_step(const bf16* __restrict__ whh,    // [dir][1024][256] bf16
               const float* __restrict__ xprojF,// [B*T][1024] f32 (natural t order)
               const float* __restrict__ xprojB,
               bf16* __restrict__ hbuf,         // [dir][2][128*256] ping-pong
               float* __restrict__ cbuf,        // [dir][128*256]
               bf16* __restrict__ out,          // [B*T][512], fwd cols 0..255, bwd 256..511
               int t, int T) {
  const int Hh = 256, G4 = 1024;
  const int dir = blockIdx.y;
  const int j0 = blockIdx.x * 32;
  const bf16* W = whh + (size_t)dir * G4 * Hh;
  const float* xp = dir ? xprojB : xprojF;
  const bf16* hprev = hbuf + ((size_t)dir * 2 + (size_t)(t & 1)) * 128 * Hh;
  bf16* hnext = hbuf + ((size_t)dir * 2 + (size_t)((t + 1) & 1)) * 128 * Hh;
  float* c = cbuf + (size_t)dir * 128 * Hh;
  const int t_eff = dir ? (T - 1 - t) : t;

  __shared__ bf16 Hs[128 * LDT];
  __shared__ bf16 Ws[128 * LDT];

  const int tid = threadIdx.x, lane = tid & 31, wv = tid >> 5;
  const int laneHalf = lane >> 4;

  // Warm L2/WGP$ for the 4 xproj gate rows this thread consumes in the
  // epilogue (global_prefetch_b8); overlaps with the recurrent GEMM below.
  {
    int b0 = wv * 16 + (laneHalf << 3);
    size_t xb0 = ((size_t)b0 * T + t_eff) * G4 + j0 + (lane & 15);
    __builtin_prefetch(&xp[xb0], 0, 1);
    __builtin_prefetch(&xp[xb0 + 256], 0, 1);
    __builtin_prefetch(&xp[xb0 + 512], 0, 1);
    __builtin_prefetch(&xp[xb0 + 768], 0, 1);
  }

  v8f acc[8];
#pragma unroll
  for (int jt = 0; jt < 8; ++jt)
#pragma unroll
    for (int v = 0; v < 8; ++v) acc[jt][v] = 0.f;

  for (int kt = 0; kt < Hh / 32; ++kt) {
#pragma unroll
    for (int i = 0; i < 2; ++i) {
      int cidx = tid + i * 256;
      int r = cidx >> 2;
      int kq = (cidx & 3) << 3;
      async_cp16(hprev + (size_t)r * Hh + (kt << 5) + kq, &Hs[r * LDT + kq]);
      int wrow = ((r >> 5) << 8) + j0 + (r & 31);  // gate-interleaved remap
      async_cp16(W + (size_t)wrow * Hh + (kt << 5) + kq, &Ws[r * LDT + kq]);
    }
    wait_async0();
    __syncthreads();
    v16bf af = load_frag(&Hs[(wv * 16) * LDT], lane);
#pragma unroll
    for (int jt = 0; jt < 8; ++jt) {
      v16bf bfg = load_frag(&Ws[(jt * 16) * LDT], lane);
      acc[jt] = wmma_bf16(af, bfg, acc[jt]);
    }
    __syncthreads();
  }

  // Gate fusion epilogue (pure registers): tile 2g+h holds gate g, jj half h.
#pragma unroll
  for (int h = 0; h < 2; ++h) {
    int jj = h * 16 + (lane & 15);
    int j = j0 + jj;
#pragma unroll
    for (int v = 0; v < 8; ++v) {
      int b = wv * 16 + (laneHalf << 3) + v;
      size_t xb = ((size_t)b * T + t_eff) * G4 + j;
      float zi = acc[0 + h][v] + xp[xb];
      float zf = acc[2 + h][v] + xp[xb + 256];
      float zg = acc[4 + h][v] + xp[xb + 512];
      float zo = acc[6 + h][v] + xp[xb + 768];
      float si = 1.f / (1.f + __expf(-zi));
      float sf = 1.f / (1.f + __expf(-zf));
      float so = 1.f / (1.f + __expf(-zo));
      float cn = sf * c[(size_t)b * Hh + j] + si * tanhf(zg);
      float hn = so * tanhf(cn);
      c[(size_t)b * Hh + j] = cn;
      hnext[(size_t)b * Hh + j] = (bf16)hn;
      out[((size_t)b * T + t_eff) * 512 + (size_t)dir * Hh + j] = (bf16)hn;
    }
  }
}

// ---------------------------------------------------------------------------
// CRF Viterbi decode: one block per batch element. trans cached in LDS.
// ---------------------------------------------------------------------------
#define KS 74
__global__ __launch_bounds__(128)
void viterbi(const float* __restrict__ em,        // [B*T][KS]
             const float* __restrict__ start_trans,
             const float* __restrict__ trans,     // [KS][KS]
             const float* __restrict__ end_trans,
             uint8_t* __restrict__ hist,          // [B][T-1][KS]
             int* __restrict__ out,               // [B][T]
             int T) {
  const int b = blockIdx.x;
  const int tid = threadIdx.x;
  __shared__ float tr[KS * KS];
  __shared__ float sc[2][KS];

  for (int i = tid; i < KS * KS; i += blockDim.x) tr[i] = trans[i];
  if (tid < KS) sc[0][tid] = start_trans[tid] + em[(size_t)b * T * KS + tid];
  __syncthreads();

  for (int t = 1; t < T; ++t) {
    int cur = t & 1, prev = cur ^ 1;
    if (tid < KS) {
      float best = -3.4e38f;
      int arg = 0;
      for (int kp = 0; kp < KS; ++kp) {
        float v = sc[prev][kp] + tr[kp * KS + tid];
        if (v > best) { best = v; arg = kp; }
      }
      sc[cur][tid] = best + em[((size_t)b * T + t) * KS + tid];
      hist[((size_t)b * (T - 1) + (t - 1)) * KS + tid] = (uint8_t)arg;
    }
    __syncthreads();
  }

  if (tid == 0) {
    int prev = (T - 1) & 1;
    float best = -3.4e38f;
    int cur = 0;
    for (int k = 0; k < KS; ++k) {
      float v = sc[prev][k] + end_trans[k];
      if (v > best) { best = v; cur = k; }
    }
    out[(size_t)b * T + (T - 1)] = cur;
    for (int t = T - 2; t >= 0; --t) {
      cur = hist[((size_t)b * (T - 1) + t) * KS + cur];
      out[(size_t)b * T + t] = cur;
    }
  }
}

// ---------------------------------------------------------------------------
// Host orchestration
// ---------------------------------------------------------------------------
extern "C" void kernel_launch(void* const* d_in, const int* in_sizes, int n_in,
                              void* d_out, int out_size, void* d_ws, size_t ws_size,
                              hipStream_t stream) {
  (void)in_sizes; (void)n_in; (void)out_size; (void)ws_size;
  const int Bq = 128, T = 256, IN = 768, HID = 512, Hh = 256, G4 = 1024, K = 74;
  const size_t M = (size_t)Bq * T;  // 32768

  const float* x = (const float*)d_in[0];
  // d_in[1] = labels (unused)
  const float *wih[2][2], *whhp[2][2], *bih[2][2], *bhh[2][2];
  int idx = 2;
  for (int l = 0; l < 2; ++l)
    for (int d = 0; d < 2; ++d) {
      wih[l][d]  = (const float*)d_in[idx++];
      whhp[l][d] = (const float*)d_in[idx++];
      bih[l][d]  = (const float*)d_in[idx++];
      bhh[l][d]  = (const float*)d_in[idx++];
    }
  const float* fcw    = (const float*)d_in[idx++];
  const float* fcb    = (const float*)d_in[idx++];
  const float* strans = (const float*)d_in[idx++];
  const float* trans  = (const float*)d_in[idx++];
  const float* etrans = (const float*)d_in[idx++];

  // ---- workspace layout (256B aligned slabs) ----
  char* wsp = (char*)d_ws;
  auto alloc = [&](size_t bytes) -> char* {
    char* p = wsp;
    wsp += (bytes + 255) & ~(size_t)255;
    return p;
  };
  bf16* xbfA    = (bf16*)alloc(M * G4 * 2);     // layer0 input (768 cols used) / layer1 output
  bf16* xbfB    = (bf16*)alloc(M * HID * 2);    // layer0 output / layer1 input
  float* xprojF = (float*)alloc(M * G4 * 4);
  float* xprojB = (float*)alloc(M * G4 * 4);
  float* em     = (float*)alloc(M * K * 4);
  uint8_t* hist = (uint8_t*)alloc((size_t)Bq * (T - 1) * K);
  bf16* hbuf    = (bf16*)alloc((size_t)2 * 2 * Bq * Hh * 2);
  float* cbuf   = (float*)alloc((size_t)2 * Bq * Hh * 4);
  bf16* wih_b[2], *whh_b[2];
  wih_b[0] = (bf16*)alloc((size_t)2 * G4 * IN * 2);   // [dir][1024][768]
  whh_b[0] = (bf16*)alloc((size_t)2 * G4 * Hh * 2);   // [dir][1024][256]
  wih_b[1] = (bf16*)alloc((size_t)2 * G4 * HID * 2);  // [dir][1024][512]
  whh_b[1] = (bf16*)alloc((size_t)2 * G4 * Hh * 2);
  bf16* fcw_b = (bf16*)alloc((size_t)K * HID * 2);

  auto cvt = [&](const float* src, bf16* dst, size_t n) {
    size_t g = (n + 255) / 256;
    if (g > 4096) g = 4096;
    cvt_f32_bf16<<<dim3((unsigned)g), 256, 0, stream>>>(src, dst, n);
  };

  // ---- convert inputs & weights to bf16 ----
  cvt(x, xbfA, M * IN);
  for (int l = 0; l < 2; ++l) {
    int insz = l ? HID : IN;
    for (int d = 0; d < 2; ++d) {
      cvt(wih[l][d],  wih_b[l] + (size_t)d * G4 * insz, (size_t)G4 * insz);
      cvt(whhp[l][d], whh_b[l] + (size_t)d * G4 * Hh,   (size_t)G4 * Hh);
    }
  }
  cvt(fcw, fcw_b, (size_t)K * HID);

  // ---- LSTM layers ----
  for (int l = 0; l < 2; ++l) {
    const bf16* inb = l ? xbfB : xbfA;
    const int insz  = l ? HID : IN;
    bf16* outb      = l ? xbfA : xbfB;

    // time-parallel input projections: xproj = in @ w_ih^T + b_ih + b_hh
    for (int d = 0; d < 2; ++d) {
      float* xpd = d ? xprojB : xprojF;
      gemm_bf16_nt<false><<<dim3((unsigned)(M / 128), G4 / 128), 256, 0, stream>>>(
          inb, insz, wih_b[l] + (size_t)d * G4 * insz, insz,
          bih[l][d], bhh[l][d], xpd, G4, (int)M, G4, insz);
    }

    // zero h ping-pong (bf16) and c (f32)
    fill_zero_u32<<<128, 256, 0, stream>>>((uint32_t*)hbuf,
                                           (size_t)2 * 2 * Bq * Hh * 2 / 4);
    fill_zero_u32<<<128, 256, 0, stream>>>((uint32_t*)cbuf, (size_t)2 * Bq * Hh);

    // sequential recurrence, fwd+bwd fused per launch
    for (int t = 0; t < T; ++t)
      lstm_step<<<dim3(8, 2), 256, 0, stream>>>(whh_b[l], xprojF, xprojB,
                                                hbuf, cbuf, outb, t, T);
  }

  // ---- emissions: em = h_final @ fc_w^T + fc_b  (final h is in xbfA) ----
  gemm_bf16_nt<true><<<dim3((unsigned)(M / 128), 1), 256, 0, stream>>>(
      xbfA, HID, fcw_b, HID, fcb, nullptr, em, K, (int)M, K, HID);

  // ---- CRF Viterbi decode ----
  viterbi<<<Bq, 128, 0, stream>>>(em, strans, trans, etrans, hist,
                                  (int*)d_out, T);
}
